// GAT_DDI_64622077935658
// MI455X (gfx1250) — matrix operand order repaired and checked by reading, not compiled, not run on hardware
//
#include <hip/hip_runtime.h>

#define NN      50000
#define NE      800000
#define NB      512
#define ND      256          // NODE_DIM == HEADS*HID
#define NH      4
#define HC      64
#define NL      3
#define ETOT    (NE + NN)    // edges + self loops

typedef __bf16 v16bf __attribute__((ext_vector_type(16)));
typedef float  v8f   __attribute__((ext_vector_type(8)));

union BF16Frag { v16bf v; uint4 q[2]; unsigned short u[16]; };

__device__ __forceinline__ unsigned short f2bf(float f) {
  unsigned u = __float_as_uint(f);
  u += 0x7fffu + ((u >> 16) & 1u);        // round-to-nearest-even
  return (unsigned short)(u >> 16);
}
__device__ __forceinline__ unsigned fkey(float f) {   // monotone float->uint
  unsigned u = __float_as_uint(f);
  return (u & 0x80000000u) ? ~u : (u | 0x80000000u);
}
__device__ __forceinline__ float funkey(unsigned k) {
  unsigned u = (k & 0x80000000u) ? (k & 0x7fffffffu) : ~k;
  return __uint_as_float(u);
}

// ---------------- weight fp32 -> bf16 ----------------
__global__ void convert_w_kernel(const float* __restrict__ Ws,
                                 unsigned short* __restrict__ Wb, int n) {
  int i = blockIdx.x * blockDim.x + threadIdx.x;
  if (i < n) Wb[i] = f2bf(Ws[i]);
}

// ---------------- WMMA GEMM: H[M,256] = X[M,256] @ Wb[256,256] ----------------
#define BM 128
#define BN 64
#define KC 64
#define LDA 72               // halfs per A row (144 B, 16B aligned)
#define LDBT 72              // halfs per B column (transposed store)

__global__ __launch_bounds__(256)
void gemm_bf16_kernel(const float* __restrict__ X,
                      const unsigned short* __restrict__ Wb,
                      float* __restrict__ H, int M) {
  __shared__ unsigned short lA[BM * LDA];        // [row][k]
  __shared__ unsigned short lBt[BN * LDBT];      // [col][k]  (transposed)
  const int tid  = threadIdx.x;
  const int lane = tid & 31;
  const int wave = tid >> 5;
  const int wm   = wave >> 1;           // 0..3  (M sub-tile of 32)
  const int wn   = wave & 1;            // 0..1  (N sub-tile of 32)
  const int bm   = blockIdx.x * BM;
  const int bn   = blockIdx.y * BN;
  const int hi   = lane >> 4;           // lane half
  const int l16  = lane & 15;
  const int kA   = hi * 8;              // A-frag K base (ISA 7.12.2, 16-bit A 16x32)
  const int kB   = hi * 8;              // B-frag K base (same K order as A frag)

  v8f acc[2][2];
  const v8f vz = {0.f,0.f,0.f,0.f,0.f,0.f,0.f,0.f};
  for (int i = 0; i < 2; i++) for (int j = 0; j < 2; j++) acc[i][j] = vz;

  const int rowA0 = wm * 32 + l16;      // A rows for tm=0 / tm=1 are +0 / +16
  const int colB0 = wn * 32 + l16;

  for (int kt = 0; kt < ND; kt += KC) {
    // stage A tile (BM x KC) f32 -> bf16 LDS, row-major
    for (int i = 0; i < 8; i++) {
      int idx = tid + i * 256;          // 0..2047, 16 float4 per row
      int row = idx >> 4;
      int c4  = (idx & 15) << 2;
      int gm  = bm + row;
      float4 v = make_float4(0.f, 0.f, 0.f, 0.f);
      if (gm < M) v = *(const float4*)(X + (size_t)gm * ND + kt + c4);
      unsigned short* p = &lA[row * LDA + c4];
      p[0] = f2bf(v.x); p[1] = f2bf(v.y); p[2] = f2bf(v.z); p[3] = f2bf(v.w);
    }
    // stage B tile (KC x BN) bf16 -> LDS transposed [col][k]
    for (int i = 0; i < 8; i++) {
      int idx  = tid + i * 256;         // 0..2047 dwords, 32 per k-row
      int row  = idx >> 5;              // k 0..63
      int col2 = (idx & 31) << 1;       // n 0..62
      unsigned d = *(const unsigned*)(Wb + (size_t)(kt + row) * ND + bn + col2);
      lBt[col2 * LDBT + row]       = (unsigned short)(d & 0xffffu);
      lBt[(col2 + 1) * LDBT + row] = (unsigned short)(d >> 16);
    }
    __syncthreads();

    for (int ks = 0; ks < KC; ks += 32) {
      BF16Frag a[2], b[2];
      for (int t = 0; t < 2; t++) {
        const unsigned short* pa = &lA[(rowA0 + t * 16) * LDA + ks + kA];
        a[t].q[0] = *(const uint4*)(pa);        // halfs K = kA..kA+7
        a[t].q[1] = *(const uint4*)(pa + 16);   // halfs K = kA+16..kA+23
      }
      for (int t = 0; t < 2; t++) {
        const unsigned short* pb = &lBt[(colB0 + t * 16) * LDBT + ks + kB];
        b[t].q[0] = *(const uint4*)(pb);        // halfs K = kB..kB+7
        b[t].q[1] = *(const uint4*)(pb + 16);   // halfs K = kB+16..kB+23
      }
      for (int i = 0; i < 2; i++)
        for (int j = 0; j < 2; j++)
          acc[i][j] = __builtin_amdgcn_wmma_f32_16x16x32_bf16(
              false, a[i].v, false, b[j].v, (short)0, acc[i][j], false, false);
    }
    __syncthreads();
  }

  // D layout: VGPR r -> (M=r, N=lane) / (M=r+8, N=lane-16)
  for (int i = 0; i < 2; i++)
    for (int j = 0; j < 2; j++)
      for (int r = 0; r < 8; r++) {
        int gm = bm + wm * 32 + i * 16 + hi * 8 + r;
        int gn = bn + wn * 32 + j * 16 + l16;
        if (gm < M) H[(size_t)gm * ND + gn] = acc[i][j][r];
      }
}

// ---------------- attention scores: asrc/adst[n,h] = <h[n,h,:], att> ----------------
__global__ void attn_kernel(const float* __restrict__ H,
                            const float* __restrict__ att_s,
                            const float* __restrict__ att_d,
                            float* __restrict__ asrc, float* __restrict__ adst) {
  int idx = blockIdx.x * blockDim.x + threadIdx.x;
  if (idx >= NN * NH) return;
  int n = idx >> 2, h = idx & 3;
  const float* hp = H + (size_t)n * ND + h * HC;
  const float* as = att_s + h * HC;
  const float* ad = att_d + h * HC;
  float s1 = 0.f, s2 = 0.f;
  for (int c = 0; c < HC; c++) { float v = hp[c]; s1 += v * as[c]; s2 += v * ad[c]; }
  asrc[idx] = s1; adst[idx] = s2;
}

__device__ __forceinline__ void edge_sd(const int* sA, const int* dA, int e, int& s, int& d) {
  if (e < NE) { s = sA[e]; d = dA[e]; } else { s = e - NE; d = s; }
}

// ---------------- segment max of leaky_relu(asrc[src]+adst[dst]) over dst ----------------
__global__ void edge_max_kernel(const int* __restrict__ sA, const int* __restrict__ dA,
                                const float* __restrict__ asrc, const float* __restrict__ adst,
                                unsigned* __restrict__ amax) {
  int idx = blockIdx.x * blockDim.x + threadIdx.x;
  if (idx >= ETOT * NH) return;
  int e = idx >> 2, h = idx & 3, s, d;
  edge_sd(sA, dA, e, s, d);
  float a = asrc[s * NH + h] + adst[d * NH + h];
  a = a > 0.f ? a : 0.2f * a;
  atomicMax(&amax[d * NH + h], fkey(a));
}

// ---------------- segment sum of exp(alpha - amax[dst]) ----------------
__global__ void edge_sum_kernel(const int* __restrict__ sA, const int* __restrict__ dA,
                                const float* __restrict__ asrc, const float* __restrict__ adst,
                                const unsigned* __restrict__ amax, float* __restrict__ denom) {
  int idx = blockIdx.x * blockDim.x + threadIdx.x;
  if (idx >= ETOT * NH) return;
  int e = idx >> 2, h = idx & 3, s, d;
  edge_sd(sA, dA, e, s, d);
  float a = asrc[s * NH + h] + adst[d * NH + h];
  a = a > 0.f ? a : 0.2f * a;
  atomicAdd(&denom[d * NH + h], __expf(a - funkey(amax[d * NH + h])));
}

// ---------------- weighted scatter: xnext[dst] += w * h[src], one wave per edge ----------------
__global__ __launch_bounds__(256)
void edge_agg_kernel(const int* __restrict__ sA, const int* __restrict__ dA,
                     const float* __restrict__ asrc, const float* __restrict__ adst,
                     const unsigned* __restrict__ amax, const float* __restrict__ denom,
                     const float* __restrict__ H, float* __restrict__ xnext) {
  int gw = (blockIdx.x * blockDim.x + threadIdx.x) >> 5;   // edge id
  int lane = threadIdx.x & 31;
  if (gw >= ETOT) return;
  int s, d;
  edge_sd(sA, dA, gw, s, d);
  float w[NH];
  for (int h = 0; h < NH; h++) {
    float a = asrc[s * NH + h] + adst[d * NH + h];
    a = a > 0.f ? a : 0.2f * a;
    float ea = __expf(a - funkey(amax[d * NH + h]));
    w[h] = ea / denom[d * NH + h];
  }
  const float* hs = H + (size_t)s * ND;
  float* xo = xnext + (size_t)d * ND;
  for (int i = 0; i < 8; i++) {
    int c = lane + i * 32;
    atomicAdd(&xo[c], w[c >> 6] * hs[c]);
  }
}

// ---------------- bias + ELU, write into xcur for next layer ----------------
__global__ void bias_elu_kernel(const float* __restrict__ xnext,
                                const float* __restrict__ bias,
                                float* __restrict__ xcur) {
  int idx = blockIdx.x * blockDim.x + threadIdx.x;
  if (idx >= NN * ND) return;
  float v = xnext[idx] + bias[idx & (ND - 1)];
  xcur[idx] = v > 0.f ? v : expm1f(v);
}

// ---------------- global_add_pool ----------------
__global__ void pool_kernel(const float* __restrict__ x, const int* __restrict__ batch,
                            float* __restrict__ pooled) {
  int idx = blockIdx.x * blockDim.x + threadIdx.x;
  if (idx >= NN * ND) return;
  int n = idx >> 8, c = idx & (ND - 1);
  atomicAdd(&pooled[(size_t)batch[n] * ND + c], x[idx]);
}

// ---------------- final MLP: relu([h|t|kge] @ W1 + b1) @ W2 + b2 ----------------
__global__ __launch_bounds__(64)
void mlp_kernel(const float* __restrict__ pH, const float* __restrict__ pT,
                const int* __restrict__ rel, const float* __restrict__ kge,
                const float* __restrict__ W1, const float* __restrict__ b1,
                const float* __restrict__ W2, const float* __restrict__ b2,
                float* __restrict__ out) {
  __shared__ float zin[2 * ND + HC];
  __shared__ float zh[HC];
  int row = blockIdx.x, t = threadIdx.x;
  for (int i = t; i < ND; i += 64) zin[i]      = pH[(size_t)row * ND + i];
  for (int i = t; i < ND; i += 64) zin[ND + i] = pT[(size_t)row * ND + i];
  zin[2 * ND + t] = kge[(size_t)rel[row] * HC + t];
  __syncthreads();
  float s = b1[t];
  for (int i = 0; i < 2 * ND + HC; i++) s += zin[i] * W1[i * HC + t];
  zh[t] = fmaxf(s, 0.f);
  __syncthreads();
  if (t == 0) {
    float o = b2[0];
    for (int j = 0; j < HC; j++) o += zh[j] * W2[j];
    out[row] = o;
  }
}

extern "C" void kernel_launch(void* const* d_in, const int* in_sizes, int n_in,
                              void* d_out, int out_size, void* d_ws, size_t ws_size,
                              hipStream_t stream) {
  const float* head_x     = (const float*)d_in[0];
  const int*   head_ei    = (const int*)d_in[1];
  const int*   head_batch = (const int*)d_in[2];
  const float* tail_x     = (const float*)d_in[3];
  const int*   tail_ei    = (const int*)d_in[4];
  const int*   tail_batch = (const int*)d_in[5];
  const int*   rel        = (const int*)d_in[6];
  const float* Ws         = (const float*)d_in[7];
  const float* att_src    = (const float*)d_in[8];
  const float* att_dst    = (const float*)d_in[9];
  const float* biases     = (const float*)d_in[10];
  const float* kge        = (const float*)d_in[11];
  const float* W1         = (const float*)d_in[12];
  const float* b1         = (const float*)d_in[13];
  const float* W2         = (const float*)d_in[14];
  const float* b2         = (const float*)d_in[15];
  float* out = (float*)d_out;

  const size_t NODE_BYTES = (size_t)NN * ND * sizeof(float);
  char* ws = (char*)d_ws;
  size_t off = 0;
  auto take = [&](size_t bytes) -> char* {
    char* p = ws + off;
    off = (off + bytes + 255) & ~(size_t)255;
    return p;
  };
  float*          xcur  = (float*)take(NODE_BYTES);
  float*          xnext = (float*)take(NODE_BYTES);
  float*          hbuf  = (float*)take(NODE_BYTES);
  unsigned short* wbf   = (unsigned short*)take((size_t)NL * ND * ND * 2);
  float*          asrc  = (float*)take((size_t)NN * NH * 4);
  float*          adst  = (float*)take((size_t)NN * NH * 4);
  unsigned*       amax  = (unsigned*)take((size_t)NN * NH * 4);
  float*          denom = (float*)take((size_t)NN * NH * 4);
  float*          poolH = (float*)take((size_t)NB * ND * 4);
  float*          poolT = (float*)take((size_t)NB * ND * 4);

  // one-time (deterministic) weight conversion
  convert_w_kernel<<<(NL * ND * ND + 255) / 256, 256, 0, stream>>>(Ws, wbf, NL * ND * ND);

  const float* xin[2]   = {head_x, tail_x};
  const int*   ei[2]    = {head_ei, tail_ei};
  const int*   batch[2] = {head_batch, tail_batch};
  float*       pooled[2] = {poolH, poolT};

  const int gAttn = (NN * NH + 255) / 256;
  const int gEdge = (ETOT * NH + 255) / 256;
  const int gAgg  = (ETOT * 32 + 255) / 256;
  const int gElem = (NN * ND + 255) / 256;

  for (int g = 0; g < 2; g++) {
    hipMemcpyAsync(xcur, xin[g], NODE_BYTES, hipMemcpyDeviceToDevice, stream);
    hipMemsetAsync(pooled[g], 0, (size_t)NB * ND * 4, stream);
    const int* sA = ei[g];
    const int* dA = ei[g] + NE;
    for (int l = 0; l < NL; l++) {
      gemm_bf16_kernel<<<dim3((NN + BM - 1) / BM, ND / BN), 256, 0, stream>>>(
          xcur, wbf + (size_t)l * ND * ND, hbuf, NN);
      attn_kernel<<<gAttn, 256, 0, stream>>>(hbuf, att_src + l * NH * HC,
                                             att_dst + l * NH * HC, asrc, adst);
      hipMemsetAsync(amax, 0, (size_t)NN * NH * 4, stream);
      hipMemsetAsync(denom, 0, (size_t)NN * NH * 4, stream);
      hipMemsetAsync(xnext, 0, NODE_BYTES, stream);
      edge_max_kernel<<<gEdge, 256, 0, stream>>>(sA, dA, asrc, adst, amax);
      edge_sum_kernel<<<gEdge, 256, 0, stream>>>(sA, dA, asrc, adst, amax, denom);
      edge_agg_kernel<<<gAgg, 256, 0, stream>>>(sA, dA, asrc, adst, amax, denom, hbuf, xnext);
      bias_elu_kernel<<<gElem, 256, 0, stream>>>(xnext, biases + l * ND, xcur);
    }
    pool_kernel<<<gElem, 256, 0, stream>>>(xcur, batch[g], pooled[g]);
  }

  mlp_kernel<<<NB, 64, 0, stream>>>(poolH, poolT, rel, kge, W1, b1, W2, b2, out);
}
